// BERT_83880711291585
// MI455X (gfx1250) — matrix operand order repaired and checked by reading, not compiled, not run on hardware
//
#include <hip/hip_runtime.h>
#include <hip/hip_bf16.h>
#include <math.h>

// ---------------------------------------------------------------------------
// BERT forward for MI455X (gfx1250): bf16 WMMA GEMMs (fp32 accumulate),
// async global->LDS staging (double buffered), fp32 elementwise epilogues.
// ---------------------------------------------------------------------------

typedef __bf16 bf16_t;
typedef bf16_t v16bf __attribute__((ext_vector_type(16)));
typedef float  v8f   __attribute__((ext_vector_type(8)));

#define TM 64            // block tile M
#define TN 64            // block tile N
#define TKK 32           // K step (matches 16x16x32 bf16 WMMA)
#define GEMM_THREADS 128 // 4 waves
#define LDSSA 36         // A LDS row stride in fp32  (144B rows, 16B multiple)
#define LDSSB 40         // B LDS row stride in bf16  (80B rows, 16B multiple)

// --- CDNA5 async global->LDS path (guarded; manual b128 fallback) ----------
#if defined(__has_builtin)
#  if __has_builtin(__builtin_amdgcn_global_load_async_to_lds_b128)
#    define HAVE_ASYNC_LDS 1
#  endif
#endif
#ifndef HAVE_ASYNC_LDS
#  define HAVE_ASYNC_LDS 0
#endif

// builtin signature (from clang diagnostic): params are pointers to a
// GCC-style vector "int __attribute__((vector_size(16)))" in addrspace(1)
// (global, printed as __device__) and addrspace(3) (LDS).
typedef int async_v4i __attribute__((vector_size(16)));
#define ASG(p) ((__attribute__((address_space(1))) async_v4i*)(p))
#define ASL(p) ((__attribute__((address_space(3))) async_v4i*)(p))

__device__ __forceinline__ void async_wait0() {
#if HAVE_ASYNC_LDS
#  if __has_builtin(__builtin_amdgcn_s_wait_asynccnt)
    __builtin_amdgcn_s_wait_asynccnt(0);
#  else
    asm volatile("s_wait_asynccnt 0" ::: "memory");
#  endif
#endif
}

// stage A tile (TM x TKK fp32) -> LDS [m][k], b128 chunks (4 per thread)
__device__ __forceinline__ void stage_A(const float* __restrict__ A, float* la,
                                        int tid, int bm, int k0, int lda)
{
    #pragma unroll
    for (int it = 0; it < (TM * TKK) / (GEMM_THREADS * 4); ++it) {
        int chunk = tid + it * GEMM_THREADS;
        int m = chunk >> 3;            // 8 x 16B chunks per 32-wide row
        int c = (chunk & 7) << 2;
        const float* gp = A + (long long)(bm + m) * lda + (k0 + c);
        float* lp = la + m * LDSSA + c;
#if HAVE_ASYNC_LDS
        __builtin_amdgcn_global_load_async_to_lds_b128(ASG(gp), ASL(lp), 0, 0);
#else
        *(float4*)lp = *(const float4*)gp;
#endif
    }
}

// stage B tile -> LDS bf16 [n][k]; float4 global loads, convert in regs
__device__ __forceinline__ void stage_B(const float* __restrict__ B, bf16_t* lb,
                                        int tid, int bn, int k0, int ldb, int transB)
{
    if (transB) {                      // B is [N][K]: rows map straight in
        #pragma unroll
        for (int it = 0; it < (TN * TKK) / (GEMM_THREADS * 4); ++it) {
            int chunk = tid + it * GEMM_THREADS;
            int nn = chunk >> 3;
            int ck = (chunk & 7) << 2;
            float4 v4 = *(const float4*)(B + (long long)(bn + nn) * ldb + (k0 + ck));
            bf16_t h4[4] = {(bf16_t)v4.x, (bf16_t)v4.y, (bf16_t)v4.z, (bf16_t)v4.w};
            *(unsigned long long*)(lb + nn * LDSSB + ck) = *(const unsigned long long*)h4;
        }
    } else {                           // B is [K][N]: transpose-scatter
        #pragma unroll
        for (int it = 0; it < (TN * TKK) / (GEMM_THREADS * 4); ++it) {
            int chunk = tid + it * GEMM_THREADS;
            int kk = chunk >> 4;       // 16 x 16B chunks per 64-wide row
            int cn = (chunk & 15) << 2;
            float4 v4 = *(const float4*)(B + (long long)(k0 + kk) * ldb + (bn + cn));
            lb[(cn + 0) * LDSSB + kk] = (bf16_t)v4.x;
            lb[(cn + 1) * LDSSB + kk] = (bf16_t)v4.y;
            lb[(cn + 2) * LDSSB + kk] = (bf16_t)v4.z;
            lb[(cn + 3) * LDSSB + kk] = (bf16_t)v4.w;
        }
    }
}

// Strided-batched GEMM: C = alpha * A * op(B) + bias
// op(B)=B (KxN) if transB==0, else B^T with B stored (NxK).
// batch z = blockIdx.z: z1=z/nz2, z2=z%nz2; operand offsets z1*s?1 + z2*s?2.
__global__ __launch_bounds__(GEMM_THREADS)
void wmma_gemm_kernel(const float* __restrict__ A, const float* __restrict__ B,
                      const float* __restrict__ bias, float* __restrict__ C,
                      int M, int N, int K, int lda, int ldb, int ldc,
                      long long sA1, long long sA2, long long sB1, long long sB2,
                      long long sC1, long long sC2, long long sb1, long long sb2,
                      float alpha, int transB, int nz2)
{
    __shared__ __align__(16) float  lAbuf[2 * TM * LDSSA];  // fp32, double buffered
    __shared__ __align__(16) bf16_t lBbuf[2 * TN * LDSSB];  // bf16 [n][k], double buffered

    const int tid  = threadIdx.x;
    const int lane = tid & 31;
    const int wave = tid >> 5;
    const int half = lane >> 4;   // 0: lanes 0-15, 1: lanes 16-31
    const int l16  = lane & 15;
    const int bm = blockIdx.y * TM;
    const int bn = blockIdx.x * TN;
    const int z  = blockIdx.z;
    const int z1 = z / nz2;
    const int z2 = z - z1 * nz2;

    A += (long long)z1 * sA1 + (long long)z2 * sA2;
    B += (long long)z1 * sB1 + (long long)z2 * sB2;
    C += (long long)z1 * sC1 + (long long)z2 * sC2;
    if (bias) bias += (long long)z1 * sb1 + (long long)z2 * sb2;

    v8f acc[4];
    #pragma unroll
    for (int t = 0; t < 4; ++t)
        #pragma unroll
        for (int r = 0; r < 8; ++r) acc[t][r] = 0.0f;

    const int NT = K / TKK;

    // prologue: stage tile 0 into buffer 0
    stage_A(A, lAbuf, tid, bm, 0, lda);
    stage_B(B, lBbuf, tid, bn, 0, ldb, transB);
    async_wait0();
    __syncthreads();

    int p = 0;
    for (int t = 0; t < NT; ++t) {
        const float*  curA = lAbuf + p * (TM * LDSSA);
        const bf16_t* curB = lBbuf + p * (TN * LDSSB);

        // stage tile t+1 into the other buffer while computing tile t
        if (t + 1 < NT) {
            stage_A(A, lAbuf + (p ^ 1) * (TM * LDSSA), tid, bm, (t + 1) * TKK, lda);
            stage_B(B, lBbuf + (p ^ 1) * (TN * LDSSB), tid, bn, (t + 1) * TKK, ldb, transB);
        }

        // A fragment (16x32): lane m=l16; elems 0..7 -> K=half*8+i,
        // elems 8..15 -> K=16+half*8+i. Two contiguous 32B fp32 runs + cvt.
        v16bf af;
        {
            const float* ap = curA + (wave * 16 + l16) * LDSSA;
            #pragma unroll
            for (int i = 0; i < 8; ++i) af[i] = (bf16_t)ap[half * 8 + i];
            #pragma unroll
            for (int i = 0; i < 8; ++i) af[8 + i] = (bf16_t)ap[16 + half * 8 + i];
        }
        // 4 x (B fragment + WMMA): lane col n=l16, K same order as A fragment
        #pragma unroll
        for (int t4 = 0; t4 < 4; ++t4) {
            const bf16_t* bp = curB + (t4 * 16 + l16) * LDSSB;
            v16bf bfr;
            #pragma unroll
            for (int i = 0; i < 16; ++i) bfr[i] = bp[(i < 8) ? (half * 8 + i) : (16 + half * 8 + (i - 8))];
            acc[t4] = __builtin_amdgcn_wmma_f32_16x16x32_bf16(
                false, af, false, bfr, (short)0, acc[t4], false, false);
        }

        async_wait0();      // this wave's async copies into buf p^1 have landed
        __syncthreads();    // publish to all waves; reads of buf p complete
        p ^= 1;
    }

    // epilogue: C/D layout: vgpr r -> m = half*8 + r, n = l16
    #pragma unroll
    for (int t = 0; t < 4; ++t) {
        int n = bn + t * 16 + l16;
        float bv = bias ? bias[n] : 0.0f;
        #pragma unroll
        for (int r = 0; r < 8; ++r) {
            int m = bm + wave * 16 + half * 8 + r;
            C[(long long)m * ldc + n] = alpha * acc[t][r] + bv;
        }
    }
}

// ---------------------------------------------------------------------------
// block reductions (blockDim == 256)
// ---------------------------------------------------------------------------
__device__ __forceinline__ float block_sum(float v, float* sm) {
    int tid = threadIdx.x;
    sm[tid] = v; __syncthreads();
    for (int s = 128; s > 0; s >>= 1) {
        if (tid < s) sm[tid] += sm[tid + s];
        __syncthreads();
    }
    float r = sm[0]; __syncthreads();
    return r;
}
__device__ __forceinline__ float block_max(float v, float* sm) {
    int tid = threadIdx.x;
    sm[tid] = v; __syncthreads();
    for (int s = 128; s > 0; s >>= 1) {
        if (tid < s) sm[tid] = fmaxf(sm[tid], sm[tid + s]);
        __syncthreads();
    }
    float r = sm[0]; __syncthreads();
    return r;
}

// embedding + positional encoding + LayerNorm (one block per token, D=768)
__global__ __launch_bounds__(256)
void embed_ln_kernel(const int* __restrict__ tok, const float* __restrict__ tokE,
                     const float* __restrict__ segE, float* __restrict__ h)
{
    __shared__ float sm[256];
    const long long row = blockIdx.x;       // b*512 + s
    const int s = (int)(row & 511);
    const int id = tok[row];
    float v[3];
    #pragma unroll
    for (int j = 0; j < 3; ++j) {
        int d = threadIdx.x + j * 256;
        float d2 = 2.0f * (float)d / 768.0f;
        float p  = (float)s * expf(-d2 * 9.210340371976184f); // / 10000^d2
        float pe = (d & 1) ? cosf(p) : sinf(p);
        v[j] = tokE[(long long)id * 768 + d] + segE[768 + d] + pe;
    }
    float mean = block_sum(v[0] + v[1] + v[2], sm) * (1.0f / 768.0f);
    float vs = 0.0f;
    #pragma unroll
    for (int j = 0; j < 3; ++j) { float t = v[j] - mean; vs += t * t; }
    float var = block_sum(vs, sm) * (1.0f / 768.0f);
    float inv = rsqrtf(var + 1e-5f);
    #pragma unroll
    for (int j = 0; j < 3; ++j) {
        int d = threadIdx.x + j * 256;
        h[row * 768 + d] = (v[j] - mean) * inv;
    }
}

// LayerNorm over D=768; y = LN(x + res) (res optional). One block per row.
__global__ __launch_bounds__(256)
void ln_kernel(const float* __restrict__ x, const float* __restrict__ res,
               float* __restrict__ y)
{
    __shared__ float sm[256];
    const long long row = blockIdx.x;
    const float* xr = x + row * 768;
    const float* rr = res ? res + row * 768 : nullptr;
    float v[3];
    #pragma unroll
    for (int j = 0; j < 3; ++j) {
        int d = threadIdx.x + j * 256;
        v[j] = xr[d] + (rr ? rr[d] : 0.0f);
    }
    float mean = block_sum(v[0] + v[1] + v[2], sm) * (1.0f / 768.0f);
    float vs = 0.0f;
    #pragma unroll
    for (int j = 0; j < 3; ++j) { float t = v[j] - mean; vs += t * t; }
    float var = block_sum(vs, sm) * (1.0f / 768.0f);
    float inv = rsqrtf(var + 1e-5f);
    #pragma unroll
    for (int j = 0; j < 3; ++j) {
        int d = threadIdx.x + j * 256;
        y[row * 768 + d] = (v[j] - mean) * inv;
    }
}

// masked softmax over rows of length 512; scores laid out [h*B+b][512][512]
__global__ __launch_bounds__(256)
void softmax_kernel(float* __restrict__ sc, const unsigned char* __restrict__ mask)
{
    __shared__ float sm[256];
    const long long row = blockIdx.x;       // (h*4+b)*512 + s
    const int z = (int)(row >> 9);
    const int b = z & 3;
    float* r = sc + row * 512;
    const unsigned char* mrow = mask + (long long)b * 512;
    const int i0 = threadIdx.x, i1 = threadIdx.x + 256;
    float v0 = mrow[i0] ? -1.0e9f : r[i0];
    float v1 = mrow[i1] ? -1.0e9f : r[i1];
    float mx = block_max(fmaxf(v0, v1), sm);
    float e0 = expf(v0 - mx), e1 = expf(v1 - mx);
    float inv = 1.0f / block_sum(e0 + e1, sm);
    r[i0] = e0 * inv;
    r[i1] = e1 * inv;
}

// exact GELU, in place
__global__ void gelu_kernel(float* __restrict__ x, long long n)
{
    long long i = (long long)blockIdx.x * 256 + threadIdx.x;
    if (i < n) {
        float v = x[i];
        x[i] = 0.5f * v * (1.0f + erff(v * 0.70710678118654752f));
    }
}

// in-place log_softmax over rows of length N (one block per row)
__global__ __launch_bounds__(256)
void log_softmax_kernel(float* __restrict__ x, int N)
{
    __shared__ float sm[256];
    float* r = x + (long long)blockIdx.x * N;
    float mx = -3.4e38f;
    for (int i = threadIdx.x; i < N; i += 256) mx = fmaxf(mx, r[i]);
    mx = block_max(mx, sm);
    float se = 0.0f;
    for (int i = threadIdx.x; i < N; i += 256) se += expf(r[i] - mx);
    se = block_sum(se, sm);
    float ls = mx + logf(se);
    for (int i = threadIdx.x; i < N; i += 256) r[i] -= ls;
}

// cls head: out[b][j] = h[b,0,:] . Wcls[:,j] + bcls[j]   (8 outputs)
__global__ void cls_kernel(const float* __restrict__ h, const float* __restrict__ Wc,
                           const float* __restrict__ bc, float* __restrict__ out)
{
    int t = threadIdx.x;
    if (t >= 8) return;
    int b = t >> 1, j = t & 1;
    const float* hr = h + (long long)b * 512 * 768;
    float s = bc[j];
    for (int d = 0; d < 768; ++d) s = fmaf(hr[d], Wc[d * 2 + j], s);
    out[b * 2 + j] = s;
}

// ---------------------------------------------------------------------------
// host side
// ---------------------------------------------------------------------------
static void launch_gemm(hipStream_t st,
    const float* A, const float* B, const float* bias, float* C,
    int M, int N, int K, int lda, int ldb, int ldc,
    long long sA1, long long sA2, long long sB1, long long sB2,
    long long sC1, long long sC2, long long sb1, long long sb2,
    float alpha, int transB, int nz1, int nz2)
{
    dim3 grid(N / TN, M / TM, nz1 * nz2);
    wmma_gemm_kernel<<<grid, GEMM_THREADS, 0, st>>>(
        A, B, bias, C, M, N, K, lda, ldb, ldc,
        sA1, sA2, sB1, sB2, sC1, sC2, sb1, sb2, alpha, transB, nz2);
}

extern "C" void kernel_launch(void* const* d_in, const int* in_sizes, int n_in,
                              void* d_out, int out_size, void* d_ws, size_t ws_size,
                              hipStream_t stream)
{
    const int*           input_tensor = (const int*)d_in[0];
    const unsigned char* attn_mask    = (const unsigned char*)d_in[1]; // bool
    const float* tok_embed = (const float*)d_in[2];
    const float* seg_embed = (const float*)d_in[3];
    const float* Wq   = (const float*)d_in[4];
    const float* bq   = (const float*)d_in[5];
    const float* Wk   = (const float*)d_in[6];
    const float* bk   = (const float*)d_in[7];
    const float* Wv   = (const float*)d_in[8];
    const float* bv   = (const float*)d_in[9];
    const float* Wo   = (const float*)d_in[10];
    const float* bo   = (const float*)d_in[11];
    const float* W1   = (const float*)d_in[12];
    const float* b1   = (const float*)d_in[13];
    const float* W2   = (const float*)d_in[14];
    const float* b2   = (const float*)d_in[15];
    const float* Wtok = (const float*)d_in[16];
    const float* btok = (const float*)d_in[17];
    const float* Wcls = (const float*)d_in[18];
    const float* bcls = (const float*)d_in[19];

    const int Dc = 768, Ec = 64, Hc = 12, Lc = 4, Vc = 32000, Ff = 3072;
    const long long BS = 2048;         // B * S
    const long long TOK = BS * Dc;     // 1,572,864

    // workspace layout (fp32)
    float* ws  = (float*)d_ws;
    float* h   = ws;                   // [2048,768]
    float* h1  = h   + TOK;
    float* t1  = h1  + TOK;
    float* q   = t1  + TOK;            // [H][2048][64]
    float* k   = q   + TOK;
    float* v   = k   + TOK;
    float* ctx = v   + TOK;            // [2048,768]
    float* ff  = ctx + TOK;            // [2048,3072]
    float* sc  = ff  + BS * Ff;        // [48][512][512]
    const long long need = (8 * TOK + BS * Ff + 48LL * 512 * 512) * 4;
    if ((long long)ws_size < need) return;

    float* out = (float*)d_out;

    // embeddings + pos + LN
    embed_ln_kernel<<<(int)BS, 256, 0, stream>>>(input_tensor, tok_embed, seg_embed, h);

    const float inv_sqrt_s = 1.0f / sqrtf(512.0f);
    const long long HDE = (long long)Hc * Dc * Ec;  // 589824
    const long long DE  = (long long)Dc * Ec;       // 49152
    const long long QH  = BS * Ec;                  // per-head stride 131072
    const long long QB  = 512LL * Ec;               // per-batch stride  32768
    const long long SCZ = 512LL * 512;              // 262144

    for (int l = 0; l < Lc; ++l) {
        // q/k/v = h @ W?[l,head] + b?[l,head]   (batched over 12 heads)
        launch_gemm(stream, h, Wq + (long long)l * HDE, bq + (long long)l * Hc * Ec, q,
                    2048, 64, 768, 768, 64, 64,
                    0, 0, DE, 0, QH, 0, Ec, 0, 1.0f, 0, 12, 1);
        launch_gemm(stream, h, Wk + (long long)l * HDE, bk + (long long)l * Hc * Ec, k,
                    2048, 64, 768, 768, 64, 64,
                    0, 0, DE, 0, QH, 0, Ec, 0, 1.0f, 0, 12, 1);
        launch_gemm(stream, h, Wv + (long long)l * HDE, bv + (long long)l * Hc * Ec, v,
                    2048, 64, 768, 768, 64, 64,
                    0, 0, DE, 0, QH, 0, Ec, 0, 1.0f, 0, 12, 1);

        // scores[h,b] = (q kT) * inv_sqrt_s    (z = h*4 + b)
        launch_gemm(stream, q, k, nullptr, sc,
                    512, 512, 64, 64, 64, 512,
                    QH, QB, QH, QB, 4 * SCZ, SCZ, 0, 0,
                    inv_sqrt_s, 1, 12, 4);

        softmax_kernel<<<48 * 512, 256, 0, stream>>>(sc, attn_mask);

        // ctx[b*512+s][h*64+e] = attn @ v   (reshape/transpose fused via ldc)
        launch_gemm(stream, sc, v, nullptr, ctx,
                    512, 64, 512, 512, 64, 768,
                    4 * SCZ, SCZ, QH, QB, 64, 512LL * 768, 0, 0,
                    1.0f, 0, 12, 4);

        // mha pre-LN: t1 = ctx @ Wo[l] + bo[l]
        launch_gemm(stream, ctx, Wo + (long long)l * Dc * Dc, bo + (long long)l * Dc, t1,
                    2048, 768, 768, 768, 768, 768,
                    0, 0, 0, 0, 0, 0, 0, 0, 1.0f, 0, 1, 1);
        ln_kernel<<<(int)BS, 256, 0, stream>>>(t1, nullptr, ctx);   // ctx = LN(t1) = mha
        ln_kernel<<<(int)BS, 256, 0, stream>>>(ctx, h, h1);         // h1 = LN(h + mha)

        // FFN
        launch_gemm(stream, h1, W1 + (long long)l * Dc * Ff, b1 + (long long)l * Ff, ff,
                    2048, 3072, 768, 768, 3072, 3072,
                    0, 0, 0, 0, 0, 0, 0, 0, 1.0f, 0, 1, 1);
        gelu_kernel<<<(int)((BS * Ff + 255) / 256), 256, 0, stream>>>(ff, BS * Ff);
        launch_gemm(stream, ff, W2 + (long long)l * Ff * Dc, b2 + (long long)l * Dc, t1,
                    2048, 768, 3072, 3072, 768, 768,
                    0, 0, 0, 0, 0, 0, 0, 0, 1.0f, 0, 1, 1);
        ln_kernel<<<(int)BS, 256, 0, stream>>>(t1, h1, h);          // h = LN(h1 + ff)
    }

    // token logits directly into d_out, then in-place log_softmax
    launch_gemm(stream, h, Wtok, btok, out,
                2048, 32000, 768, 768, 32000, 32000,
                0, 0, 0, 0, 0, 0, 0, 0, 1.0f, 0, 1, 1);
    log_softmax_kernel<<<(int)BS, 256, 0, stream>>>(out, Vc);

    // cls head appended after token_logp
    cls_kernel<<<1, 32, 0, stream>>>(h, Wcls, bcls, out + BS * (long long)Vc);
}
